// MGSTGNN_45853070852226
// MI455X (gfx1250) — matrix operand order repaired and checked by reading, not compiled, not run on hardware
//
#include <hip/hip_runtime.h>
#include <math.h>

// MGSTGNN for MI455X (gfx1250): wave32 + v_wmma_f32_16x16x32_f16 tensor path,
// vectorized LDS fragments (ds_load_b128) and async global->LDS tile staging.
// B=16, T=12, N=450, D=3, H=64, E=16, GH=256, MH=256, P=2, OS=12, OD=1

#define NB 16
#define NT 12
#define NN 450
#define ND 3
#define NH 64
#define NE 16
#define NGH 256
#define NN2 900
#define NEGBIG (-9.0e15f)

typedef __attribute__((ext_vector_type(16))) _Float16 v16h;
typedef __attribute__((ext_vector_type(8)))  float    v8f;
typedef __attribute__((ext_vector_type(2)))  _Float16 h2v;

__device__ inline v8f wmma_16x16x32(v16h a, v16h b, v8f c) {
  return __builtin_amdgcn_wmma_f32_16x16x32_f16(false, a, false, b, (short)0, c,
                                                false, false);
}

// K index for A tiles (identity, contiguous order matching B tiles).
// lanes 0-15 read halves [0..15]  = K {0..15}
// lanes 16-31 read halves [16..31] = K {16..31}
__device__ inline int permA(int k) {
  return k;
}

// Shared fragment loader for tiles stored as [row][32] halfs where each
// lane reads 16 contiguous halves at byte offset {0,32}: one 32B LDS load.
__device__ inline v16h frag_ld16(const _Float16* tile) {
  int lane = threadIdx.x & 31;
  const _Float16* p = tile + (lane & 15) * 32 + ((lane < 16) ? 0 : 16);
  return *(const v16h*)p;
}

// Async global->LDS copy (CDNA5 ASYNCcnt path). LDS offset = low 32 bits of
// the generic address.
__device__ inline void async_copy_b128(void* lds, const void* gaddr) {
  unsigned l = (unsigned)(size_t)lds;
  asm volatile("global_load_async_to_lds_b128 %0, %1, off"
               :: "v"(l), "v"(gaddr)
               : "memory");
}
__device__ inline void wait_async0() {
  asm volatile("s_wait_asynccnt 0" ::: "memory");
}

template <int ACT>
__device__ inline float act_fn(float x) {
  if constexpr (ACT == 1) {           // exact gelu
    return 0.5f * x * (1.0f + erff(x * 0.70710678118654752f));
  } else if constexpr (ACT == 2) {    // elu(abs(x)) == abs(x)
    return fabsf(x);
  } else {
    return x;
  }
}

// ---------------------------------------------------------------------------
// Generic batched GEMM: C[M,N] = act(A[M,K] @ B[K,N] + bias[N])
// Block 128 threads (4 waves), tile 32(M) x 64(N) x 32(K).
// TRANSB: B stored as [N,K] row-major.
// ---------------------------------------------------------------------------
template <int ACT, bool TRANSB>
__global__ void gemm_wmma(const float* __restrict__ A, const float* __restrict__ B,
                          const float* __restrict__ bias, float* __restrict__ C,
                          int M, int N, int K, long sA, long sB, long sC) {
  __shared__ _Float16 As[32][32];    // [m][permA(k)]
  __shared__ _Float16 Bst[64][32];   // [n][k] (transposed)
  const int bz = blockIdx.z;
  A += (size_t)bz * sA;
  B += (size_t)bz * sB;
  C += (size_t)bz * sC;
  const int m0 = blockIdx.y * 32;
  const int n0 = blockIdx.x * 64;
  const int tid = threadIdx.x;
  const int wave = tid >> 5;
  const int waveM = wave & 1;
  const int waveN = wave >> 1;
  v8f acc0 = {0.f, 0.f, 0.f, 0.f, 0.f, 0.f, 0.f, 0.f};
  v8f acc1 = acc0;

  for (int ks = 0; ks < K; ks += 32) {
    // A tile: pair-packed stores into permuted layout
    for (int e = tid; e < 512; e += 128) {
      int r = e >> 4, c = (e & 15) * 2;
      int gm = m0 + r;
      float v0 = 0.f, v1 = 0.f;
      if (gm < M) {
        const float* ap = A + (size_t)gm * K + ks + c;
        if (ks + c < K)     v0 = ap[0];
        if (ks + c + 1 < K) v1 = ap[1];
      }
      h2v hv; hv[0] = (_Float16)v0; hv[1] = (_Float16)v1;
      *(h2v*)&As[r][permA(c)] = hv;
    }
    // B tile: transposed, pair-packed along K
    for (int e = tid; e < 1024; e += 128) {
      int n = e & 63, k2 = (e >> 6) * 2;
      int gn = n0 + n;
      int gk0 = ks + k2, gk1 = gk0 + 1;
      float v0 = 0.f, v1 = 0.f;
      if (gn < N) {
        if (TRANSB) {
          const float* bp = B + (size_t)gn * K;
          if (gk0 < K) v0 = bp[gk0];
          if (gk1 < K) v1 = bp[gk1];
        } else {
          if (gk0 < K) v0 = B[(size_t)gk0 * N + gn];
          if (gk1 < K) v1 = B[(size_t)gk1 * N + gn];
        }
      }
      h2v hv; hv[0] = (_Float16)v0; hv[1] = (_Float16)v1;
      *(h2v*)&Bst[n][k2] = hv;
    }
    // speculative prefetch of next A tile row (global_prefetch_b8)
    if (ks + 32 < K) {
      int pr = m0 + (tid & 31);
      if (pr < M) __builtin_prefetch(&A[(size_t)pr * K + ks + 32], 0, 1);
    }
    __syncthreads();
    v16h af = frag_ld16(&As[waveM * 16][0]);
    v16h b0 = frag_ld16(&Bst[waveN * 32][0]);
    v16h b1 = frag_ld16(&Bst[waveN * 32 + 16][0]);
    acc0 = wmma_16x16x32(af, b0, acc0);
    acc1 = wmma_16x16x32(af, b1, acc1);
    __syncthreads();
  }

  const int lane = tid & 31;
  const int rbase = m0 + waveM * 16 + ((lane < 16) ? 0 : 8);
  const int cbase = n0 + waveN * 32 + (lane & 15);
#pragma unroll
  for (int s = 0; s < 2; ++s) {
    v8f acc = s ? acc1 : acc0;
    int gc = cbase + s * 16;
    if (gc >= N) continue;
    float bv = bias ? bias[gc] : 0.f;
#pragma unroll
    for (int v = 0; v < 8; ++v) {
      int gr = rbase + v;
      if (gr < M) C[(size_t)gr * N + gc] = act_fn<ACT>(acc[v] + bv);
    }
  }
}

// ---------------------------------------------------------------------------
// LayerNorm of (node_emb + time_emb[t]) for all (inst, t, gate): 108 x 450 rows
// ---------------------------------------------------------------------------
__global__ void en_kernel(const float* __restrict__ nemb, const float* __restrict__ temb,
                          const float* __restrict__ lnw0, const float* __restrict__ lnb0,
                          const float* __restrict__ lnws, const float* __restrict__ lnbs,
                          float* __restrict__ EnAll) {
  int idx = blockIdx.x * blockDim.x + threadIdx.x;
  if (idx >= 3 * NT * 3 * NN) return;
  int n = idx % NN;
  int r = idx / NN;
  int g = r % 3; r /= 3;
  int t = r % NT;
  int inst = r / NT;
  const float* w;
  const float* bb;
  if (inst == 0) { w = lnw0 + g * NE; bb = lnb0 + g * NE; }
  else { w = lnws + ((inst - 1) * 3 + g) * NE; bb = lnbs + ((inst - 1) * 3 + g) * NE; }
  float x[NE];
  float mu = 0.f;
#pragma unroll
  for (int e = 0; e < NE; ++e) {
    x[e] = nemb[n * NE + e] + temb[t * NE + e];
    mu += x[e];
  }
  mu *= (1.f / NE);
  float var = 0.f;
#pragma unroll
  for (int e = 0; e < NE; ++e) { float d = x[e] - mu; var += d * d; }
  var *= (1.f / NE);
  float inv = rsqrtf(var + 1e-12f);
#pragma unroll
  for (int e = 0; e < NE; ++e)
    EnAll[(size_t)idx * NE + e] = (x[e] - mu) * inv * w[e] + bb[e];
}

// ---------------------------------------------------------------------------
// aff[g] = softmax(En_g @ En_g^T, axis=1). grid (450, 3), block 256.
// ---------------------------------------------------------------------------
__global__ void aff_kernel(const float* __restrict__ En, float* __restrict__ aff) {
  __shared__ float s[NN];
  __shared__ float red[256];
  __shared__ float En_n[NE];
  int n = blockIdx.x, g = blockIdx.y, tid = threadIdx.x;
  const float* Eg = En + (size_t)g * NN * NE;
  if (tid < NE) En_n[tid] = Eg[(size_t)n * NE + tid];
  __syncthreads();
  float mx = -3.0e38f;
  for (int m = tid; m < NN; m += 256) {
    float d = 0.f;
#pragma unroll
    for (int e = 0; e < NE; ++e) d += En_n[e] * Eg[(size_t)m * NE + e];
    s[m] = d;
    mx = fmaxf(mx, d);
  }
  red[tid] = mx; __syncthreads();
  for (int st = 128; st > 0; st >>= 1) { if (tid < st) red[tid] = fmaxf(red[tid], red[tid + st]); __syncthreads(); }
  float gmax = red[0]; __syncthreads();
  float sum = 0.f;
  for (int m = tid; m < NN; m += 256) { float v = expf(s[m] - gmax); s[m] = v; sum += v; }
  red[tid] = sum; __syncthreads();
  for (int st = 128; st > 0; st >>= 1) { if (tid < st) red[tid] += red[tid + st]; __syncthreads(); }
  float inv = 1.f / red[0];
  __syncthreads();
  for (int m = tid; m < NN; m += 256)
    aff[((size_t)g * NN + n) * NN + m] = s[m] * inv;
}

// ---------------------------------------------------------------------------
// dst[b,n,:] = [ x(b,n,0..xdim-1), (z?z*state:state)(b,n,0..63) ]
// ---------------------------------------------------------------------------
__global__ void build_input(float* __restrict__ dst, const float* __restrict__ xsrc,
                            long xbs, long xns, int xdim,
                            const float* __restrict__ state, const float* __restrict__ z) {
  int cd = xdim + NH;
  int tot = NB * NN * cd;
  int idx = blockIdx.x * blockDim.x + threadIdx.x;
  if (idx >= tot) return;
  int c = idx % cd;
  int n = (idx / cd) % NN;
  int b = idx / (cd * NN);
  float v;
  if (c < xdim) {
    v = xsrc[(size_t)b * xbs + (size_t)n * xns + c];
  } else {
    size_t si = ((size_t)b * NN + n) * NH + (c - xdim);
    v = state[si];
    if (z) v *= z[si];
  }
  dst[idx] = v;
}

// ---------------------------------------------------------------------------
// Per-node GCN apply: out[b,n,o] = post( sum_c xg[b,n,c]*(En[n]·wpool[:,c,o]) + En[n]·bpool[:,o] )
// xg = [p0 | p1] along c (c in [0,2*Din)). Block = one node, 128 thr (4 waves).
// POST 0: dst = sigmoid(x).  POST 1: state = r*state + (1-r)*tanh(x).
// ---------------------------------------------------------------------------
template <int POST>
__global__ void gcn_apply(const float* __restrict__ p0, const float* __restrict__ p1,
                          const float* __restrict__ En, const float* __restrict__ wpool,
                          const float* __restrict__ bpool, int Din,
                          float* __restrict__ dst, const float* __restrict__ rbuf,
                          float* __restrict__ state) {
  __shared__ float Ens[NE];
  __shared__ _Float16 As[16][32];    // [b][permA(c)]
  __shared__ _Float16 Wst[64][32];   // [o][kc] (transposed weight tile)
  const int n = blockIdx.x;
  const int tid = threadIdx.x;
  const int wave = tid >> 5;
  if (tid < NE) Ens[tid] = En[(size_t)n * NE + tid];
  __syncthreads();
  const int C2 = 2 * Din;
  const size_t dstride = (size_t)C2 * NH;
  v8f acc = {0.f, 0.f, 0.f, 0.f, 0.f, 0.f, 0.f, 0.f};

  for (int ks = 0; ks < C2; ks += 32) {
    // A tile: 16 batch rows x 32 channels of xg (pair-packed, permuted)
    for (int e = tid; e < 256; e += 128) {
      int bb = e >> 4, c = (e & 15) * 2;
      size_t ro = ((size_t)bb * NN + n) * Din;
      float v0 = 0.f, v1 = 0.f;
      int g0 = ks + c, g1 = g0 + 1;
      if (g0 < C2) v0 = (g0 < Din) ? p0[ro + g0] : p1[ro + g0 - Din];
      if (g1 < C2) v1 = (g1 < Din) ? p0[ro + g1] : p1[ro + g1 - Din];
      h2v hv; hv[0] = (_Float16)v0; hv[1] = (_Float16)v1;
      *(h2v*)&As[bb][permA(c)] = hv;
    }
    // W tile: generate per-node weights on the fly, transposed pair store
    for (int e = tid; e < 1024; e += 128) {
      int o = e & 63, k2 = (e >> 6) * 2;
      float v0 = 0.f, v1 = 0.f;
      int g0 = ks + k2, g1 = g0 + 1;
      if (g0 < C2) {
        const float* wp = wpool + (size_t)g0 * NH + o;
#pragma unroll
        for (int d = 0; d < NE; ++d) v0 += Ens[d] * wp[d * dstride];
      }
      if (g1 < C2) {
        const float* wp = wpool + (size_t)g1 * NH + o;
#pragma unroll
        for (int d = 0; d < NE; ++d) v1 += Ens[d] * wp[d * dstride];
      }
      h2v hv; hv[0] = (_Float16)v0; hv[1] = (_Float16)v1;
      *(h2v*)&Wst[o][k2] = hv;
    }
    __syncthreads();
    v16h af = frag_ld16(&As[0][0]);
    v16h bf = frag_ld16(&Wst[wave * 16][0]);
    acc = wmma_16x16x32(af, bf, acc);
    __syncthreads();
  }

  const int lane = tid & 31;
  const int o = wave * 16 + (lane & 15);
  float bias = 0.f;
#pragma unroll
  for (int d = 0; d < NE; ++d) bias += Ens[d] * bpool[d * NH + o];
  const int bb0 = (lane < 16) ? 0 : 8;
#pragma unroll
  for (int v = 0; v < 8; ++v) {
    int bb = bb0 + v;
    size_t idx = ((size_t)bb * NN + n) * NH + o;
    float x = acc[v] + bias;
    if constexpr (POST == 0) {
      dst[idx] = 1.f / (1.f + expf(-x));
    } else {
      float hc = tanhf(x);
      float r = rbuf[idx];
      float s = state[idx];
      state[idx] = r * s + (1.f - r) * hc;
    }
  }
}

// ---------------------------------------------------------------------------
// Wh[b,i,:] = h[b,i,:] @ W (K=3), plus Wh1 = Wh@a[:GH], Wh2 = Wh@a[GH:]
// ---------------------------------------------------------------------------
__global__ void wh_kernel(const float* __restrict__ src, const float* __restrict__ W,
                          const float* __restrict__ a, float* __restrict__ Wh,
                          float* __restrict__ Wh1, float* __restrict__ Wh2, int t) {
  __shared__ float red[256];
  int i = blockIdx.x, b = blockIdx.y, o = threadIdx.x;
  int tprev = (t == 0) ? 0 : (t - 1);
  int tt = (i < NN) ? tprev : t;
  int nn = (i < NN) ? i : (i - NN);
  const float* h = src + (((size_t)b * NT + tt) * NN + nn) * ND;
  float wh = h[0] * W[o] + h[1] * W[NGH + o] + h[2] * W[2 * NGH + o];
  Wh[((size_t)b * NN2 + i) * NGH + o] = wh;
  red[o] = wh * a[o];
  __syncthreads();
  for (int st = 128; st > 0; st >>= 1) { if (o < st) red[o] += red[o + st]; __syncthreads(); }
  if (o == 0) Wh1[(size_t)b * NN2 + i] = red[0];
  __syncthreads();
  red[o] = wh * a[NGH + o];
  __syncthreads();
  for (int st = 128; st > 0; st >>= 1) { if (o < st) red[o] += red[o + st]; __syncthreads(); }
  if (o == 0) Wh2[(size_t)b * NN2 + i] = red[0];
}

// ---------------------------------------------------------------------------
// Softmax(axis=1) column stats: Mc[b,j] = max_i m(i,j), Sc = sum_i exp(m-Mc)
// ---------------------------------------------------------------------------
__global__ void colstats_kernel(const float* __restrict__ Wh1, const float* __restrict__ Wh2,
                                const float* __restrict__ adj,
                                float* __restrict__ Mc, float* __restrict__ Sc) {
  __shared__ float red[256];
  int j = blockIdx.x, b = blockIdx.y, tid = threadIdx.x;
  float w2 = Wh2[(size_t)b * NN2 + j];
  float mx = -3.0e38f;
  for (int i = tid; i < NN2; i += 256) {
    float e = Wh1[(size_t)b * NN2 + i] + w2;
    e = (e > 0.f) ? e : 0.2f * e;
    float m = (adj[(size_t)i * NN2 + j] > 0.f) ? e : NEGBIG;
    mx = fmaxf(mx, m);
  }
  red[tid] = mx; __syncthreads();
  for (int st = 128; st > 0; st >>= 1) { if (tid < st) red[tid] = fmaxf(red[tid], red[tid + st]); __syncthreads(); }
  float gmax = red[0]; __syncthreads();
  float sum = 0.f;
  for (int i = tid; i < NN2; i += 256) {
    float e = Wh1[(size_t)b * NN2 + i] + w2;
    e = (e > 0.f) ? e : 0.2f * e;
    float m = (adj[(size_t)i * NN2 + j] > 0.f) ? e : NEGBIG;
    sum += expf(m - gmax);
  }
  red[tid] = sum; __syncthreads();
  for (int st = 128; st > 0; st >>= 1) { if (tid < st) red[tid] += red[tid + st]; __syncthreads(); }
  if (tid == 0) { Mc[(size_t)b * NN2 + j] = gmax; Sc[(size_t)b * NN2 + j] = red[0]; }
}

// ---------------------------------------------------------------------------
// hp2 = elu(reshape(att @ Wh)): attention generated on the fly (never in HBM),
// Wh tiles staged with async global->LDS b128 copies (ASYNCcnt path).
// grid (4, 29, 16), block 128.
// ---------------------------------------------------------------------------
__global__ void gat_attn_kernel(const float* __restrict__ Wh, const float* __restrict__ Wh1,
                                const float* __restrict__ Wh2, const float* __restrict__ Mc,
                                const float* __restrict__ Sc, const float* __restrict__ adj,
                                float* __restrict__ hp2) {
  __shared__ _Float16 As[32][32];    // [i][permA(j)] attention tile
  __shared__ float    Braw[32][64];  // async landing zone (f32)
  __shared__ _Float16 Bst[64][32];   // [n][k] converted/transposed
  const int b = blockIdx.z;
  const int i0 = blockIdx.y * 32, o0 = blockIdx.x * 64;
  const int tid = threadIdx.x;
  const int wave = tid >> 5, waveM = wave & 1, waveN = wave >> 1;
  v8f acc0 = {0.f, 0.f, 0.f, 0.f, 0.f, 0.f, 0.f, 0.f};
  v8f acc1 = acc0;
  const float* wh1 = Wh1 + (size_t)b * NN2;
  const float* wh2 = Wh2 + (size_t)b * NN2;
  const float* mc = Mc + (size_t)b * NN2;
  const float* sc = Sc + (size_t)b * NN2;
  const float* whb = Wh + (size_t)b * NN2 * NGH;

  for (int ks = 0; ks < NN2; ks += 32) {
    // A tile: attention weights computed on the fly (pair-packed, permuted)
    for (int e = tid; e < 512; e += 128) {
      int r = e >> 4, c = (e & 15) * 2;
      int gi = i0 + r;
      float v0 = 0.f, v1 = 0.f;
      if (gi < NN2) {
#pragma unroll
        for (int q = 0; q < 2; ++q) {
          int gj = ks + c + q;
          if (gj < NN2) {
            float ee = wh1[gi] + wh2[gj];
            ee = (ee > 0.f) ? ee : 0.2f * ee;
            float m = (adj[(size_t)gi * NN2 + gj] > 0.f) ? ee : NEGBIG;
            float av = expf(m - mc[gj]) / sc[gj];
            if (q == 0) v0 = av; else v1 = av;
          }
        }
      }
      h2v hv; hv[0] = (_Float16)v0; hv[1] = (_Float16)v1;
      *(h2v*)&As[r][permA(c)] = hv;
    }
    // B tile: Wh rows. Full tiles use async global->LDS, then convert.
    if (ks + 32 <= NN2) {
      for (int e = tid; e < 512; e += 128) {
        int r = e >> 4, cq = (e & 15) * 4;
        async_copy_b128(&Braw[r][cq], whb + (size_t)(ks + r) * NGH + o0 + cq);
      }
      wait_async0();
      __syncthreads();
      for (int e = tid; e < 1024; e += 128) {
        int n = e & 63, k2 = (e >> 6) * 2;
        h2v hv;
        hv[0] = (_Float16)Braw[k2][n];
        hv[1] = (_Float16)Braw[k2 + 1][n];
        *(h2v*)&Bst[n][k2] = hv;
      }
    } else {
      for (int e = tid; e < 1024; e += 128) {
        int n = e & 63, k2 = (e >> 6) * 2;
        float v0 = (ks + k2 < NN2)     ? whb[(size_t)(ks + k2) * NGH + o0 + n]     : 0.f;
        float v1 = (ks + k2 + 1 < NN2) ? whb[(size_t)(ks + k2 + 1) * NGH + o0 + n] : 0.f;
        h2v hv; hv[0] = (_Float16)v0; hv[1] = (_Float16)v1;
        *(h2v*)&Bst[n][k2] = hv;
      }
    }
    __syncthreads();
    v16h af = frag_ld16(&As[waveM * 16][0]);
    v16h b0 = frag_ld16(&Bst[waveN * 32][0]);
    v16h b1 = frag_ld16(&Bst[waveN * 32 + 16][0]);
    acc0 = wmma_16x16x32(af, b0, acc0);
    acc1 = wmma_16x16x32(af, b1, acc1);
    __syncthreads();
  }

  const int lane = tid & 31;
  const int rbase = i0 + waveM * 16 + ((lane < 16) ? 0 : 8);
  const int cbase = o0 + waveN * 32 + (lane & 15);
#pragma unroll
  for (int s = 0; s < 2; ++s) {
    v8f acc = s ? acc1 : acc0;
    int gc = cbase + s * 16;
#pragma unroll
    for (int v = 0; v < 8; ++v) {
      int gi = rbase + v;
      if (gi < NN2) {
        int n = (gi < NN) ? gi : gi - NN;
        int c = gc + ((gi < NN) ? 0 : NGH);
        float x = acc[v];
        hp2[((size_t)b * NN + n) * (2 * NGH) + c] = (x > 0.f) ? x : expm1f(x);
      }
    }
  }
}

// inner[t] = LN(state + att_out), eps 1e-5. One thread per (b,n).
__global__ void addln_kernel(const float* __restrict__ s, const float* __restrict__ a,
                             const float* __restrict__ w, const float* __restrict__ bb,
                             float* __restrict__ o) {
  int idx = blockIdx.x * blockDim.x + threadIdx.x;
  if (idx >= NB * NN) return;
  size_t base = (size_t)idx * NH;
  float sum = 0.f, sq = 0.f;
  for (int h = 0; h < NH; ++h) { float v = s[base + h] + a[base + h]; sum += v; sq += v * v; }
  float mu = sum * (1.f / NH);
  float var = sq * (1.f / NH) - mu * mu;
  float inv = rsqrtf(var + 1e-5f);
  for (int h = 0; h < NH; ++h) {
    float v = s[base + h] + a[base + h];
    o[base + h] = (v - mu) * inv * w[h] + bb[h];
  }
}

// Final: LN(states1)/LN(base) (eps 1e-12) then 12-way conv over (P=2, H=64).
__global__ void final_kernel(const float* __restrict__ s1, const float* __restrict__ bs,
                             const float* __restrict__ onw, const float* __restrict__ onb,
                             const float* __restrict__ cw, const float* __restrict__ cb,
                             float* __restrict__ out) {
  __shared__ float vsh[2][NH];
  __shared__ float red[NH];
  int bn = blockIdx.x;
  int b = bn / NN, n = bn % NN;
  int h = threadIdx.x;
  size_t base = (size_t)bn * NH;
  for (int p = 0; p < 2; ++p) {
    float v = (p == 0) ? s1[base + h] : bs[base + h];
    red[h] = v; __syncthreads();
    for (int st = 32; st > 0; st >>= 1) { if (h < st) red[h] += red[h + st]; __syncthreads(); }
    float mu = red[0] * (1.f / NH); __syncthreads();
    float d = v - mu;
    red[h] = d * d; __syncthreads();
    for (int st = 32; st > 0; st >>= 1) { if (h < st) red[h] += red[h + st]; __syncthreads(); }
    float var = red[0] * (1.f / NH); __syncthreads();
    vsh[p][h] = d * rsqrtf(var + 1e-12f) * onw[h] + onb[h];
  }
  __syncthreads();
  if (h < 12) {
    float acc = cb[h];
    for (int p = 0; p < 2; ++p)
      for (int hh = 0; hh < NH; ++hh)
        acc += vsh[p][hh] * cw[(h * 2 + p) * NH + hh];
    out[((size_t)b * 12 + h) * NN + n] = acc;
  }
}

__global__ void zero_kernel(float* __restrict__ p, int n) {
  int i = blockIdx.x * blockDim.x + threadIdx.x;
  if (i < n) p[i] = 0.f;
}

// ---------------------------------------------------------------------------
extern "C" void kernel_launch(void* const* d_in, const int* in_sizes, int n_in,
                              void* d_out, int out_size, void* d_ws, size_t ws_size,
                              hipStream_t stream) {
  const float* src   = (const float*)d_in[0];
  const float* adj   = (const float*)d_in[1];
  const float* nemb  = (const float*)d_in[2];
  const float* temb  = (const float*)d_in[3];
  const float* g0w   = (const float*)d_in[4];
  const float* g0b   = (const float*)d_in[5];
  const float* g0lnw = (const float*)d_in[6];
  const float* g0lnb = (const float*)d_in[7];
  const float* gsw   = (const float*)d_in[8];
  const float* gsb   = (const float*)d_in[9];
  const float* gslnw = (const float*)d_in[10];
  const float* gslnb = (const float*)d_in[11];
  const float* gatW  = (const float*)d_in[12];
  const float* gatA  = (const float*)d_in[13];
  const float* f1w   = (const float*)d_in[14];
  const float* f1b   = (const float*)d_in[15];
  const float* f2w   = (const float*)d_in[16];
  const float* f2b   = (const float*)d_in[17];
  const float* ntw   = (const float*)d_in[18];
  const float* ntb   = (const float*)d_in[19];
  const float* onw   = (const float*)d_in[20];
  const float* onb   = (const float*)d_in[21];
  const float* cw    = (const float*)d_in[22];
  const float* cb    = (const float*)d_in[23];
  float* out = (float*)d_out;
  (void)in_sizes; (void)n_in; (void)out_size; (void)ws_size;

  float* base = (float*)d_ws;
  size_t off = 0;
  auto alloc = [&](size_t n) {
    float* p = base + off;
    off += (n + 63) & ~(size_t)63;
    return p;
  };
  float* EnAll = alloc((size_t)3 * NT * 3 * NN * NE);     // 108 LN'd embeddings
  float* aff   = alloc((size_t)3 * NN * NN);              // per-step gate affinities
  float* inp   = alloc((size_t)NB * NN * 128);
  float* cand  = alloc((size_t)NB * NN * 128);
  float* xz    = alloc((size_t)NB * NN * 128);
  float* xr    = alloc((size_t)NB * NN * 128);
  float* xc    = alloc((size_t)NB * NN * 128);
  float* zb    = alloc((size_t)NB * NN * NH);
  float* rb    = alloc((size_t)NB * NN * NH);
  float* st    = alloc((size_t)NB * NN * NH);             // gru0 state -> base
  float* s0    = alloc((size_t)NB * NN * NH);
  float* s1    = alloc((size_t)NB * NN * NH);
  float* inner = alloc((size_t)NT * NB * NN * NH);
  float* Wh    = alloc((size_t)NB * NN2 * NGH);
  float* Wh1   = alloc((size_t)NB * NN2);
  float* Wh2   = alloc((size_t)NB * NN2);
  float* Mc    = alloc((size_t)NB * NN2);
  float* Sc    = alloc((size_t)NB * NN2);
  float* hp2   = alloc((size_t)NB * NN * 2 * NGH);
  float* y1    = alloc((size_t)NB * NN * 256);
  float* ao    = alloc((size_t)NB * NN * NH);

  const int stateElems = NB * NN * NH;
  en_kernel<<<(3 * NT * 3 * NN + 255) / 256, 256, 0, stream>>>(
      nemb, temb, g0lnw, g0lnb, gslnw, gslnb, EnAll);
  zero_kernel<<<(stateElems + 255) / 256, 256, 0, stream>>>(st, stateElems);
  zero_kernel<<<(stateElems + 255) / 256, 256, 0, stream>>>(s0, stateElems);
  zero_kernel<<<(stateElems + 255) / 256, 256, 0, stream>>>(s1, stateElems);

  // ---------------- Loop 1: gru0 + GAT ----------------
  for (int t = 0; t < NT; ++t) {
    const int Din = ND + NH;                               // 67
    const size_t wgate = (size_t)NE * 2 * Din * NH;        // per-gate wpool elems
    const float* EnB = EnAll + ((size_t)t * 3) * NN * NE;  // inst 0
    const float* xsrc = src + (size_t)t * NN * ND;
    const long xbs = (long)NT * NN * ND, xns = ND;
    const long sBC = (long)NN * Din;

    aff_kernel<<<dim3(NN, 3), 256, 0, stream>>>(EnB, aff);
    build_input<<<(NB * NN * Din + 255) / 256, 256, 0, stream>>>(
        inp, xsrc, xbs, xns, ND, st, nullptr);
    dim3 gAgg((Din + 63) / 64, (NN + 31) / 32, NB);
    gemm_wmma<0, false><<<gAgg, 128, 0, stream>>>(
        aff + (size_t)1 * NN * NN, inp, nullptr, xz, NN, Din, NN, 0L, sBC, sBC);
    gemm_wmma<0, false><<<gAgg, 128, 0, stream>>>(
        aff + (size_t)0 * NN * NN, inp, nullptr, xr, NN, Din, NN, 0L, sBC, sBC);
    gcn_apply<0><<<NN, 128, 0, stream>>>(
        inp, xz, EnB + 1 * NN * NE, g0w + 1 * wgate, g0b + 1 * NE * NH, Din,
        zb, nullptr, nullptr);
    gcn_apply<0><<<NN, 128, 0, stream>>>(
        inp, xr, EnB + 0 * NN * NE, g0w + 0 * wgate, g0b + 0 * NE * NH, Din,
        rb, nullptr, nullptr);
    build_input<<<(NB * NN * Din + 255) / 256, 256, 0, stream>>>(
        cand, xsrc, xbs, xns, ND, st, zb);
    gemm_wmma<0, false><<<gAgg, 128, 0, stream>>>(
        aff + (size_t)2 * NN * NN, cand, nullptr, xc, NN, Din, NN, 0L, sBC, sBC);
    gcn_apply<1><<<NN, 128, 0, stream>>>(
        cand, xc, EnB + 2 * NN * NE, g0w + 2 * wgate, g0b + 2 * NE * NH, Din,
        nullptr, rb, st);

    // GAT branch
    wh_kernel<<<dim3(NN2, NB), 256, 0, stream>>>(
        src, gatW + (size_t)t * ND * NGH, gatA + (size_t)t * 2 * NGH,
        Wh, Wh1, Wh2, t);
    colstats_kernel<<<dim3(NN2, NB), 256, 0, stream>>>(Wh1, Wh2, adj, Mc, Sc);
    gat_attn_kernel<<<dim3(NGH / 64, (NN2 + 31) / 32, NB), 128, 0, stream>>>(
        Wh, Wh1, Wh2, Mc, Sc, adj, hp2);
    gemm_wmma<1, true><<<dim3(256 / 64, (NN + 31) / 32, NB), 128, 0, stream>>>(
        hp2, f1w + (size_t)t * 256 * 512, f1b + t * 256, y1,
        NN, 256, 512, (long)NN * 512, 0L, (long)NN * 256);
    gemm_wmma<2, true><<<dim3(1, (NN + 31) / 32, NB), 128, 0, stream>>>(
        y1, f2w + (size_t)t * NH * 256, f2b + t * NH, ao,
        NN, NH, 256, (long)NN * 256, 0L, (long)NN * NH);
    addln_kernel<<<(NB * NN + 255) / 256, 256, 0, stream>>>(
        st, ao, ntw + t * NH, ntb + t * NH, inner + (size_t)t * stateElems);
  }

  // ---------------- Loop 2: alternating grus ----------------
  for (int t = 0; t < NT; ++t) {
    const int i = t & 1;
    const int Din = 2 * NH;                                // 128
    const size_t wgate = (size_t)NE * 2 * Din * NH;
    float* sts = i ? s1 : s0;
    const float* EnB = EnAll + ((size_t)((1 + i) * NT + t) * 3) * NN * NE;
    const float* xsrc = inner + (size_t)t * stateElems;
    const long xbs = (long)NN * NH, xns = NH;
    const long sBC = (long)NN * Din;

    aff_kernel<<<dim3(NN, 3), 256, 0, stream>>>(EnB, aff);
    build_input<<<(NB * NN * Din + 255) / 256, 256, 0, stream>>>(
        inp, xsrc, xbs, xns, NH, sts, nullptr);
    dim3 gAgg((Din + 63) / 64, (NN + 31) / 32, NB);
    gemm_wmma<0, false><<<gAgg, 128, 0, stream>>>(
        aff + (size_t)1 * NN * NN, inp, nullptr, xz, NN, Din, NN, 0L, sBC, sBC);
    gemm_wmma<0, false><<<gAgg, 128, 0, stream>>>(
        aff + (size_t)0 * NN * NN, inp, nullptr, xr, NN, Din, NN, 0L, sBC, sBC);
    gcn_apply<0><<<NN, 128, 0, stream>>>(
        inp, xz, EnB + 1 * NN * NE, gsw + ((size_t)i * 3 + 1) * wgate,
        gsb + ((size_t)i * 3 + 1) * NE * NH, Din, zb, nullptr, nullptr);
    gcn_apply<0><<<NN, 128, 0, stream>>>(
        inp, xr, EnB + 0 * NN * NE, gsw + ((size_t)i * 3 + 0) * wgate,
        gsb + ((size_t)i * 3 + 0) * NE * NH, Din, rb, nullptr, nullptr);
    build_input<<<(NB * NN * Din + 255) / 256, 256, 0, stream>>>(
        cand, xsrc, xbs, xns, NH, sts, zb);
    gemm_wmma<0, false><<<gAgg, 128, 0, stream>>>(
        aff + (size_t)2 * NN * NN, cand, nullptr, xc, NN, Din, NN, 0L, sBC, sBC);
    gcn_apply<1><<<NN, 128, 0, stream>>>(
        cand, xc, EnB + 2 * NN * NE, gsw + ((size_t)i * 3 + 2) * wgate,
        gsb + ((size_t)i * 3 + 2) * NE * NH, Din, nullptr, rb, sts);
  }

  // out[:, -2:] of stack([s0, s1, base]) == (s1, base)
  final_kernel<<<NB * NN, NH, 0, stream>>>(s1, st, onw, onb, cw, cb, out);
}
